// Transformer_66649302499713
// MI455X (gfx1250) — matrix operand order repaired and checked by reading, not compiled, not run on hardware
//
#include <hip/hip_runtime.h>
#include <cstddef>
#include <cstdint>

// ---------------------------------------------------------------------------
// MI455X (gfx1250) transformer forward, v3.
//  * f16 everywhere on the GEMM path (weights transpose-converted once),
//    tiles staged by the Tensor Data Mover (tensor_load_to_lds), double-
//    buffered LDS, s_wait_tensorcnt pipelining, v_wmma_f32_16x16x32_f16.
//  * v3: 2x2 register blocking per wave (32x32), block tile 128x64:
//    8 LDS fragment loads feed 4 WMMAs (was 6 loads for 2).
// ---------------------------------------------------------------------------

typedef _Float16 v16h __attribute__((ext_vector_type(16)));
typedef _Float16 v8h  __attribute__((ext_vector_type(8)));
typedef _Float16 v4h  __attribute__((ext_vector_type(4)));
typedef float    v8f  __attribute__((ext_vector_type(8)));
typedef unsigned int u32x4 __attribute__((ext_vector_type(4)));
typedef int      i32x4 __attribute__((ext_vector_type(4)));
typedef int      i32x8 __attribute__((ext_vector_type(8)));

#define LDST 72            // LDS row stride in halves (144B): 64 data + 8 pad
#define OPA  (128 * LDST)  // A tile halves (128 rows)
#define OPB  (64 * LDST)   // B tile halves (64 rows)
#define BUFH (OPA + OPB)   // halves per double-buffer slot

#if defined(__HIP_DEVICE_COMPILE__) && __has_builtin(__builtin_amdgcn_tensor_load_to_lds)
#define HAVE_TDM 1
#else
#define HAVE_TDM 0
#endif

union Frag16 { v16h v; v8h h[2]; };

#if HAVE_TDM
// TDM 2D tile load: `rows` rows x 128B (64 f16) from global (row stride
// ldHalves) into LDS at byte offset ldsOff; +4 dwords pad per 32 dwords so
// LDS rows land on the 72-half stride. D# bitfields per CDNA5 ISA ch.8.
__device__ __forceinline__ void tdm_tile(unsigned ldsOff, const _Float16* g,
                                         int ldHalves, unsigned rows)
{
    const unsigned long long ga = (unsigned long long)(uintptr_t)g;
    u32x4 g0;
    g0[0] = 1u;                                   // count=1, user descriptor
    g0[1] = ldsOff;                               // lds_addr (bytes)
    g0[2] = (unsigned)ga;                         // global_addr[31:0]
    g0[3] = (unsigned)(ga >> 32) | (2u << 30);    // global_addr[56:32] | type=2
    const unsigned td0 = 0x40000000u;             // huge tensor dims: no OOB clip
    const unsigned td1 = 0x40000000u;
    const unsigned tile0 = 16u;                   // 16 x 8B = 128B row
    const unsigned str0 = (unsigned)(ldHalves >> 2); // row stride in 8B units
    i32x8 g1;
    g1[0] = (int)((3u << 16) | (1u << 20) | (4u << 22) | (3u << 25));
    //        data_size=8B    pad_enable   interval=32dw  amount=4dw
    g1[1] = (int)(td0 << 16);                     // tensor_dim0[15:0]
    g1[2] = (int)((td0 >> 16) | (td1 << 16));     // tensor_dim0[31:16] | dim1 lo
    g1[3] = (int)((td1 >> 16) | (tile0 << 16));   // dim1 hi | tile_dim0
    g1[4] = (int)rows;                            // tile_dim1 (tile_dim2=0)
    g1[5] = (int)str0;                            // tensor_dim0_stride[31:0]
    g1[6] = 0;
    g1[7] = 0;
    const i32x4 z4 = {0, 0, 0, 0};
#if defined(__clang_major__) && (__clang_major__ >= 23)
    const i32x8 z8 = {0, 0, 0, 0, 0, 0, 0, 0};
    __builtin_amdgcn_tensor_load_to_lds(g0, g1, z4, z4, z8, 0);
#else
    __builtin_amdgcn_tensor_load_to_lds(g0, g1, z4, z4, 0);
#endif
}
#endif

// C = scale * A[M,K] @ B[N,K]^T + bias.  A,B f16; accumulate f32.
// OUTMODE: 0 = f32 C ; 1 = f16 row-major C16 ; 2 = f16 transposed C16 [N][M].
// Block 256 thr = 8 waves (4 M-bands x 2 N-bands); 128x64 C tile; wave tile
// 32x32 as 2x2 WMMA accumulators; BK=64 (2 WMMA K-steps per staged tile).
template<int OUTMODE, int RELU>
__global__ __launch_bounds__(256)
void gemm_f16_kernel(const _Float16* __restrict__ A, int lda, long sAb, long sAh,
                     const _Float16* __restrict__ Bm, int ldb, long sBb, long sBh,
                     float* __restrict__ Cf, int ldc, long sCb, long sCh,
                     _Float16* __restrict__ C16, int ldc16, long sC16b, long sC16h,
                     const float* __restrict__ bias, int nH, float scale, int Ktiles)
{
    extern __shared__ _Float16 smem[];   // [2][A 128x72 | B 64x72]

    const int tid  = threadIdx.x;
    const int lane = tid & 31;
    const int wave = tid >> 5;
    const int wm   = (wave & 3) * 32;    // M band
    const int wn   = (wave >> 2) * 32;   // N band

    const int z  = blockIdx.z;
    const int bo = z / nH;
    const int hh = z - bo * nH;

    const _Float16* Ab = A  + (size_t)bo * sAb + (size_t)hh * sAh;
    const _Float16* Bb = Bm + (size_t)bo * sBb + (size_t)hh * sBh;

    const int gm = blockIdx.y * 128;
    const int gn = blockIdx.x * 64;

    const _Float16* Atile = Ab + (size_t)gm * lda;   // + kt*64 per step
    const _Float16* Btile = Bb + (size_t)gn * ldb;

    v8f acc00 = {}, acc01 = {}, acc10 = {}, acc11 = {};

    // ISA wave32 fragment maps:
    //  A 16x32: M=lane&15 ; elems 0-7 K=kbA.. , elems 8-15 K=16+kbA..
    //  B 32x16: N=lane&15 ; elems 0-7 K=kbB.. , elems 8-15 K=16+kbB..
    const int mrow = lane & 15;
    const int kbA  = (lane >> 4) * 8;
    const int kbB  = (lane >> 4) * 8;

#if HAVE_TDM
    if (tid < 32) {                       // wave 0 drives the DMA pipeline
        tdm_tile(0u,       Atile, lda, 128u);
        tdm_tile(OPA * 2u, Btile, ldb, 64u);
    }
#endif
    for (int kt = 0; kt < Ktiles; ++kt) {
        const _Float16* bufA = smem + (size_t)(kt & 1) * BUFH;
        const _Float16* bufB = bufA + OPA;

#if HAVE_TDM
        if (tid < 32) {
            if (kt + 1 < Ktiles) {        // prefetch next tile into other buffer
                const unsigned nb = (unsigned)((kt + 1) & 1) * (BUFH * 2u);
                tdm_tile(nb,            Atile + (size_t)(kt + 1) * 64, lda, 128u);
                tdm_tile(nb + OPA * 2u, Btile + (size_t)(kt + 1) * 64, ldb, 64u);
                __builtin_amdgcn_s_wait_tensorcnt((short)2);   // tile kt complete
            } else {
                __builtin_amdgcn_s_wait_tensorcnt((short)0);
            }
        }
        __syncthreads();
#else
        // Fallback: cooperative copy, fixed trip counts so loads batch.
        {
            const size_t ko = (size_t)kt * 64;
            _Float16* dA = const_cast<_Float16*>(bufA);
            _Float16* dB = const_cast<_Float16*>(bufB);
#pragma unroll
            for (int i = 0; i < 4; ++i) {               // A: 1024 chunks of 8h
                const int c = tid + i * 256;
                const int r = c >> 3, s = (c & 7) << 3;
                v8h vch = *reinterpret_cast<const v8h*>(Atile + (size_t)r * lda + ko + s);
                *reinterpret_cast<v8h*>(dA + r * LDST + s) = vch;
            }
#pragma unroll
            for (int i = 0; i < 2; ++i) {               // B: 512 chunks of 8h
                const int c = tid + i * 256;
                const int r = c >> 3, s = (c & 7) << 3;
                v8h vch = *reinterpret_cast<const v8h*>(Btile + (size_t)r * ldb + ko + s);
                *reinterpret_cast<v8h*>(dB + r * LDST + s) = vch;
            }
        }
        __syncthreads();
#endif

#pragma unroll
        for (int kk = 0; kk < 2; ++kk) {
            const int ko = kk * 32;
            Frag16 a0, a1, b0, b1;
            a0.h[0] = *reinterpret_cast<const v8h*>(&bufA[(wm + mrow) * LDST + ko + kbA]);
            a0.h[1] = *reinterpret_cast<const v8h*>(&bufA[(wm + mrow) * LDST + ko + 16 + kbA]);
            a1.h[0] = *reinterpret_cast<const v8h*>(&bufA[(wm + 16 + mrow) * LDST + ko + kbA]);
            a1.h[1] = *reinterpret_cast<const v8h*>(&bufA[(wm + 16 + mrow) * LDST + ko + 16 + kbA]);
            b0.h[0] = *reinterpret_cast<const v8h*>(&bufB[(wn + mrow) * LDST + ko + kbB]);
            b0.h[1] = *reinterpret_cast<const v8h*>(&bufB[(wn + mrow) * LDST + ko + 16 + kbB]);
            b1.h[0] = *reinterpret_cast<const v8h*>(&bufB[(wn + 16 + mrow) * LDST + ko + kbB]);
            b1.h[1] = *reinterpret_cast<const v8h*>(&bufB[(wn + 16 + mrow) * LDST + ko + 16 + kbB]);
            acc00 = __builtin_amdgcn_wmma_f32_16x16x32_f16(false, a0.v, false, b0.v,
                                                           (short)0, acc00, false, false);
            acc01 = __builtin_amdgcn_wmma_f32_16x16x32_f16(false, a0.v, false, b1.v,
                                                           (short)0, acc01, false, false);
            acc10 = __builtin_amdgcn_wmma_f32_16x16x32_f16(false, a1.v, false, b0.v,
                                                           (short)0, acc10, false, false);
            acc11 = __builtin_amdgcn_wmma_f32_16x16x32_f16(false, a1.v, false, b1.v,
                                                           (short)0, acc11, false, false);
        }
        __syncthreads();
    }

    // ---- epilogue: C/D map -> N = lane&15, M = r + 8*(lane>>4) --------------
    const int   nc0 = gn + wn + mrow;
    const int   nc1 = nc0 + 16;
    const float bv0 = bias ? bias[nc0] : 0.0f;
    const float bv1 = bias ? bias[nc1] : 0.0f;
    const int   rsh = (lane >> 4) * 8;
    const int   m0  = gm + wm + rsh;       // rows of acc*0 ; +16 for acc1*

    if (OUTMODE == 0) {
        float* Cb = Cf + (size_t)bo * sCb + (size_t)hh * sCh;
#pragma unroll
        for (int r = 0; r < 8; ++r) {
            float o00 = acc00[r] * scale + bv0;
            float o01 = acc01[r] * scale + bv1;
            float o10 = acc10[r] * scale + bv0;
            float o11 = acc11[r] * scale + bv1;
            if (RELU) {
                o00 = fmaxf(o00, 0.0f); o01 = fmaxf(o01, 0.0f);
                o10 = fmaxf(o10, 0.0f); o11 = fmaxf(o11, 0.0f);
            }
            Cb[(size_t)(m0 + r)      * ldc + nc0] = o00;
            Cb[(size_t)(m0 + r)      * ldc + nc1] = o01;
            Cb[(size_t)(m0 + 16 + r) * ldc + nc0] = o10;
            Cb[(size_t)(m0 + 16 + r) * ldc + nc1] = o11;
        }
    } else if (OUTMODE == 1) {
        _Float16* Cb = C16 + (size_t)bo * sC16b + (size_t)hh * sC16h;
#pragma unroll
        for (int r = 0; r < 8; ++r) {
            float o00 = acc00[r] * scale + bv0;
            float o01 = acc01[r] * scale + bv1;
            float o10 = acc10[r] * scale + bv0;
            float o11 = acc11[r] * scale + bv1;
            if (RELU) {
                o00 = fmaxf(o00, 0.0f); o01 = fmaxf(o01, 0.0f);
                o10 = fmaxf(o10, 0.0f); o11 = fmaxf(o11, 0.0f);
            }
            Cb[(size_t)(m0 + r)      * ldc16 + nc0] = (_Float16)o00;
            Cb[(size_t)(m0 + r)      * ldc16 + nc1] = (_Float16)o01;
            Cb[(size_t)(m0 + 16 + r) * ldc16 + nc0] = (_Float16)o10;
            Cb[(size_t)(m0 + 16 + r) * ldc16 + nc1] = (_Float16)o11;
        }
    } else {   // OUTMODE == 2: transposed f16, 16B stores along M
        _Float16* Cb = C16 + (size_t)bo * sC16b + (size_t)hh * sC16h;
        v8h p00, p01, p10, p11;
#pragma unroll
        for (int r = 0; r < 8; ++r) {
            p00[r] = (_Float16)(acc00[r] * scale + bv0);
            p01[r] = (_Float16)(acc01[r] * scale + bv1);
            p10[r] = (_Float16)(acc10[r] * scale + bv0);
            p11[r] = (_Float16)(acc11[r] * scale + bv1);
        }
        *reinterpret_cast<v8h*>(&Cb[(size_t)nc0 * ldc16 + m0])      = p00;
        *reinterpret_cast<v8h*>(&Cb[(size_t)nc0 * ldc16 + m0 + 16]) = p10;
        *reinterpret_cast<v8h*>(&Cb[(size_t)nc1 * ldc16 + m0])      = p01;
        *reinterpret_cast<v8h*>(&Cb[(size_t)nc1 * ldc16 + m0 + 16]) = p11;
    }
}

// softmax over 512-long rows: f32 scores in, f16 probabilities out.
__global__ __launch_bounds__(256)
void softmax_kernel(const float* __restrict__ sc, _Float16* __restrict__ out)
{
    __shared__ float red[256];
    const size_t roff = (size_t)blockIdx.x * 512;
    const int tid = threadIdx.x;
    const float a0 = sc[roff + tid], a1 = sc[roff + tid + 256];

    red[tid] = fmaxf(a0, a1); __syncthreads();
    for (int s = 128; s > 0; s >>= 1) {
        if (tid < s) red[tid] = fmaxf(red[tid], red[tid + s]);
        __syncthreads();
    }
    const float m = red[0]; __syncthreads();

    const float e0 = __expf(a0 - m), e1 = __expf(a1 - m);
    red[tid] = e0 + e1; __syncthreads();
    for (int s = 128; s > 0; s >>= 1) {
        if (tid < s) red[tid] += red[tid + s];
        __syncthreads();
    }
    const float inv = 1.0f / red[0];
    out[roff + tid]       = (_Float16)(e0 * inv);
    out[roff + tid + 256] = (_Float16)(e1 * inv);
}

// out = LayerNorm(a + b) * g + be, dual f32 + f16 outputs; rows of 512.
__global__ __launch_bounds__(256)
void add_ln_kernel(const float* __restrict__ a, const float* __restrict__ b,
                   const float* __restrict__ g, const float* __restrict__ be,
                   float* __restrict__ out, _Float16* __restrict__ out16)
{
    __shared__ float red[256];
    const size_t roff = (size_t)blockIdx.x * 512;
    const int tid = threadIdx.x;

    const float x0 = a[roff + tid]       + b[roff + tid];
    const float x1 = a[roff + tid + 256] + b[roff + tid + 256];

    red[tid] = x0 + x1; __syncthreads();
    for (int s = 128; s > 0; s >>= 1) {
        if (tid < s) red[tid] += red[tid + s];
        __syncthreads();
    }
    const float mu = red[0] * (1.0f / 512.0f); __syncthreads();

    const float d0 = x0 - mu, d1 = x1 - mu;
    red[tid] = d0 * d0 + d1 * d1; __syncthreads();
    for (int s = 128; s > 0; s >>= 1) {
        if (tid < s) red[tid] += red[tid + s];
        __syncthreads();
    }
    const float rstd = rsqrtf(red[0] * (1.0f / 512.0f) + 1e-5f);

    const float o0 = d0 * rstd * g[tid]       + be[tid];
    const float o1 = d1 * rstd * g[tid + 256] + be[tid + 256];
    out[roff + tid]         = o0;
    out[roff + tid + 256]   = o1;
    out16[roff + tid]       = (_Float16)o0;
    out16[roff + tid + 256] = (_Float16)o1;
}

// f32 [K,N] -> f16 [N,K] tiled transpose-convert; blockIdx.z = layer.
__global__ __launch_bounds__(256)
void transpose_cvt_kernel(const float* __restrict__ in, _Float16* __restrict__ out,
                          int K, int N)
{
    __shared__ float t[32][33];
    const size_t lofs = (size_t)blockIdx.z * K * N;
    const int k0 = blockIdx.y * 32, n0 = blockIdx.x * 32;
    in  += lofs;
    out += lofs;
#pragma unroll
    for (int j = 0; j < 32; j += 8)
        t[threadIdx.y + j][threadIdx.x] =
            in[(size_t)(k0 + threadIdx.y + j) * N + n0 + threadIdx.x];
    __syncthreads();
#pragma unroll
    for (int j = 0; j < 32; j += 8)
        out[(size_t)(n0 + threadIdx.y + j) * K + k0 + threadIdx.x] =
            (_Float16)t[threadIdx.x][threadIdx.y + j];
}

// elementwise f32 -> f16 (n multiple of 1024)
__global__ __launch_bounds__(256)
void cvt16_kernel(const float* __restrict__ in, _Float16* __restrict__ out)
{
    const size_t i = ((size_t)blockIdx.x * 256 + threadIdx.x) * 4;
    const float4 f = *reinterpret_cast<const float4*>(in + i);
    v4h hv = { (_Float16)f.x, (_Float16)f.y, (_Float16)f.z, (_Float16)f.w };
    *reinterpret_cast<v4h*>(out + i) = hv;
}

// -------------------------- host-side orchestration -------------------------

enum { OUT_F32 = 0, OUT_F16 = 1, OUT_F16T = 2 };

static void gemm(hipStream_t st, int mode, bool relu,
                 const _Float16* A, int lda, long sAb, long sAh,
                 const _Float16* B, int ldb, long sBb, long sBh,
                 float* Cf, int ldc, long sCb, long sCh,
                 _Float16* C16, int ldc16, long sC16b, long sC16h,
                 const float* bias, int M, int N, int K,
                 int batches, int nH, float scale)
{
    dim3 grid(N / 64, M / 128, batches), block(256);
    const int kt = K / 64;
    const size_t sh = (size_t)BUFH * 2 * sizeof(_Float16);   // 54 KB double buffer
    if (mode == OUT_F32)
        gemm_f16_kernel<0, 0><<<grid, block, sh, st>>>(A, lda, sAb, sAh, B, ldb, sBb, sBh,
            Cf, ldc, sCb, sCh, C16, ldc16, sC16b, sC16h, bias, nH, scale, kt);
    else if (mode == OUT_F16 && relu)
        gemm_f16_kernel<1, 1><<<grid, block, sh, st>>>(A, lda, sAb, sAh, B, ldb, sBb, sBh,
            Cf, ldc, sCb, sCh, C16, ldc16, sC16b, sC16h, bias, nH, scale, kt);
    else if (mode == OUT_F16)
        gemm_f16_kernel<1, 0><<<grid, block, sh, st>>>(A, lda, sAb, sAh, B, ldb, sBb, sBh,
            Cf, ldc, sCb, sCh, C16, ldc16, sC16b, sC16h, bias, nH, scale, kt);
    else
        gemm_f16_kernel<2, 0><<<grid, block, sh, st>>>(A, lda, sAb, sAh, B, ldb, sBb, sBh,
            Cf, ldc, sCb, sCh, C16, ldc16, sC16b, sC16h, bias, nH, scale, kt);
}

static void transpose_cvt(hipStream_t st, const float* in, _Float16* out,
                          int K, int N, int layers)
{
    dim3 grid(N / 32, K / 32, layers), block(32, 8);
    transpose_cvt_kernel<<<grid, block, 0, st>>>(in, out, K, N);
}

// MHA with f16 operands; writes f32 projection output (pre-LayerNorm).
static void attention_block(hipStream_t st,
                            const _Float16* xq16, const _Float16* xkv16,
                            const _Float16* WqT, const float* bq,
                            const _Float16* WkT, const float* bk,
                            const _Float16* WvT, const float* bv,
                            const _Float16* WoT, const float* bo,
                            _Float16* q16, _Float16* k16, _Float16* vT16,
                            float* sc, _Float16* p16, _Float16* attn16,
                            float* out_f32)
{
    const int M = 2048, E = 512, S = 512, Hh = 8, Dh = 64;
    const long sBE = (long)S * E, sSS = (long)S * S;

    gemm(st, OUT_F16, false, xq16, E, 0, 0, WqT, E, 0, 0,
         nullptr, 0, 0, 0, q16, E, 0, 0, bq, M, E, E, 1, 1, 1.0f);
    gemm(st, OUT_F16, false, xkv16, E, 0, 0, WkT, E, 0, 0,
         nullptr, 0, 0, 0, k16, E, 0, 0, bk, M, E, E, 1, 1, 1.0f);
    gemm(st, OUT_F16T, false, xkv16, E, 0, 0, WvT, E, 0, 0,
         nullptr, 0, 0, 0, vT16, M, 0, 0, bv, M, E, E, 1, 1, 1.0f);   // V^T [E][M]

    // scores[b,h] = (1/8) Q_bh K_bh^T  (f32)
    gemm(st, OUT_F32, false, q16, E, sBE, Dh, k16, E, sBE, Dh,
         sc, S, (long)Hh * sSS, sSS, nullptr, 0, 0, 0, nullptr,
         S, S, Dh, 4 * Hh, Hh, 0.125f);

    softmax_kernel<<<dim3(4 * Hh * S), dim3(256), 0, st>>>(sc, p16);

    // ctx[b,h] = P_bh @ (V^T)_bh^T  -> f16 row-major into 64-col slabs
    gemm(st, OUT_F16, false, p16, S, (long)Hh * sSS, sSS,
         vT16, M, 512, (long)Dh * M,
         nullptr, 0, 0, 0, attn16, E, sBE, Dh, nullptr,
         S, Dh, S, 4 * Hh, Hh, 1.0f);

    gemm(st, OUT_F32, false, attn16, E, 0, 0, WoT, E, 0, 0,
         out_f32, E, 0, 0, nullptr, 0, 0, 0, bo, M, E, E, 1, 1, 1.0f);
}

extern "C" void kernel_launch(void* const* d_in, const int* in_sizes, int n_in,
                              void* d_out, int out_size, void* d_ws, size_t ws_size,
                              hipStream_t stream)
{
    (void)in_sizes; (void)n_in; (void)out_size; (void)ws_size;

    constexpr int Bb = 4, S = 512, E = 512, Lnum = 6, FF = 2048, Vv = 16000;
    const size_t BSE = (size_t)Bb * S * E;   // 1,048,576
    const int    M   = Bb * S;               // 2048

    const float* src = (const float*)d_in[1];
    const float* trg = (const float*)d_in[3];

    // Param dicts flattened in jax-pytree (alphabetical key) order.
    const float* eW1 = (const float*)d_in[5];
    const float* eW2 = (const float*)d_in[6];
    const float* eWk = (const float*)d_in[7];
    const float* eWo = (const float*)d_in[8];
    const float* eWq = (const float*)d_in[9];
    const float* eWv = (const float*)d_in[10];
    const float* ebk = (const float*)d_in[11];
    const float* ebo = (const float*)d_in[12];
    const float* ebq = (const float*)d_in[13];
    const float* ebv = (const float*)d_in[14];
    const float* ec1 = (const float*)d_in[15];
    const float* ec2 = (const float*)d_in[16];
    const float* ed1 = (const float*)d_in[17];
    const float* ed2 = (const float*)d_in[18];
    const float* eg1 = (const float*)d_in[19];
    const float* eg2 = (const float*)d_in[20];
    const float* dW1  = (const float*)d_in[21];
    const float* dW2  = (const float*)d_in[22];
    const float* dWk  = (const float*)d_in[23];
    const float* dWk2 = (const float*)d_in[24];
    const float* dWo  = (const float*)d_in[25];
    const float* dWo2 = (const float*)d_in[26];
    const float* dWq  = (const float*)d_in[27];
    const float* dWq2 = (const float*)d_in[28];
    const float* dWv  = (const float*)d_in[29];
    const float* dWv2 = (const float*)d_in[30];
    const float* dbk  = (const float*)d_in[31];
    const float* dbk2 = (const float*)d_in[32];
    const float* dbo  = (const float*)d_in[33];
    const float* dbo2 = (const float*)d_in[34];
    const float* dbq  = (const float*)d_in[35];
    const float* dbq2 = (const float*)d_in[36];
    const float* dbv  = (const float*)d_in[37];
    const float* dbv2 = (const float*)d_in[38];
    const float* dc1  = (const float*)d_in[39];
    const float* dc2  = (const float*)d_in[40];
    const float* dd1  = (const float*)d_in[41];
    const float* dd2  = (const float*)d_in[42];
    const float* dd3  = (const float*)d_in[43];
    const float* dg1  = (const float*)d_in[44];
    const float* dg2  = (const float*)d_in[45];
    const float* dg3  = (const float*)d_in[46];
    const float* Wout = (const float*)d_in[47];
    const float* bout = (const float*)d_in[48];

    // -------- workspace carve: f32 region then f16 region --------
    float* fp = (float*)d_ws;
    float* sc  = fp; fp += 8 * BSE;   // scores f32 [B,H,S,S]
    float* tmp = fp; fp += BSE;
    float* v1  = fp; fp += BSE;
    float* b2  = fp; fp += BSE;
    float* x   = fp; fp += BSE;       // current activations f32

    _Float16* hp = (_Float16*)fp;
    _Float16* x16    = hp; hp += BSE;
    _Float16* v116   = hp; hp += BSE;
    _Float16* b216   = hp; hp += BSE;
    _Float16* q16    = hp; hp += BSE;
    _Float16* k16    = hp; hp += BSE;
    _Float16* vT16   = hp; hp += BSE;       // V^T  [E][M]
    _Float16* attn16 = hp; hp += BSE;
    _Float16* enc16  = hp; hp += BSE;
    _Float16* h16    = hp; hp += 4 * BSE;   // FFN hidden f16 [M,FF]
    _Float16* p16    = hp; hp += 8 * BSE;   // probs f16 [B,H,S,S]

    const size_t EE = (size_t)E * E, EF = (size_t)E * FF;
    _Float16* eWqT = hp; hp += Lnum * EE;
    _Float16* eWkT = hp; hp += Lnum * EE;
    _Float16* eWvT = hp; hp += Lnum * EE;
    _Float16* eWoT = hp; hp += Lnum * EE;
    _Float16* eW1T = hp; hp += Lnum * EF;   // [FF][E] per layer
    _Float16* eW2T = hp; hp += Lnum * EF;   // [E][FF] per layer
    _Float16* dWqT  = hp; hp += Lnum * EE;
    _Float16* dWkT  = hp; hp += Lnum * EE;
    _Float16* dWvT  = hp; hp += Lnum * EE;
    _Float16* dWoT  = hp; hp += Lnum * EE;
    _Float16* dWq2T = hp; hp += Lnum * EE;
    _Float16* dWk2T = hp; hp += Lnum * EE;
    _Float16* dWv2T = hp; hp += Lnum * EE;
    _Float16* dWo2T = hp; hp += Lnum * EE;
    _Float16* dW1T  = hp; hp += Lnum * EF;
    _Float16* dW2T  = hp; hp += Lnum * EF;
    _Float16* WoutT = hp; hp += (size_t)E * Vv;   // [V][E]

    // -------- one-time weight transpose-converts (f32 -> f16 [N][K]) --------
    transpose_cvt(stream, eWq, eWqT, E, E, Lnum);
    transpose_cvt(stream, eWk, eWkT, E, E, Lnum);
    transpose_cvt(stream, eWv, eWvT, E, E, Lnum);
    transpose_cvt(stream, eWo, eWoT, E, E, Lnum);
    transpose_cvt(stream, eW1, eW1T, E, FF, Lnum);
    transpose_cvt(stream, eW2, eW2T, FF, E, Lnum);
    transpose_cvt(stream, dWq,  dWqT,  E, E, Lnum);
    transpose_cvt(stream, dWk,  dWkT,  E, E, Lnum);
    transpose_cvt(stream, dWv,  dWvT,  E, E, Lnum);
    transpose_cvt(stream, dWo,  dWoT,  E, E, Lnum);
    transpose_cvt(stream, dWq2, dWq2T, E, E, Lnum);
    transpose_cvt(stream, dWk2, dWk2T, E, E, Lnum);
    transpose_cvt(stream, dWv2, dWv2T, E, E, Lnum);
    transpose_cvt(stream, dWo2, dWo2T, E, E, Lnum);
    transpose_cvt(stream, dW1, dW1T, E, FF, Lnum);
    transpose_cvt(stream, dW2, dW2T, FF, E, Lnum);
    transpose_cvt(stream, Wout, WoutT, E, Vv, 1);

    // ---------------- encoder ----------------
    cvt16_kernel<<<dim3((unsigned)(BSE / 1024)), dim3(256), 0, stream>>>(src, x16);
    for (int l = 0; l < Lnum; ++l) {
        const size_t wEE = (size_t)l * EE, wE = (size_t)l * E;
        const size_t wEF = (size_t)l * EF, wF = (size_t)l * FF;
        const float* xres = (l == 0) ? src : x;

        attention_block(stream, x16, x16,
                        eWqT + wEE, ebq + wE, eWkT + wEE, ebk + wE,
                        eWvT + wEE, ebv + wE, eWoT + wEE, ebo + wE,
                        q16, k16, vT16, sc, p16, attn16, tmp);
        add_ln_kernel<<<dim3(M), dim3(256), 0, stream>>>(xres, tmp, eg1 + wE, ed1 + wE, v1, v116);

        gemm(stream, OUT_F16, true, v116, E, 0, 0, eW1T + wEF, E, 0, 0,
             nullptr, 0, 0, 0, h16, FF, 0, 0, ec1 + wF, M, FF, E, 1, 1, 1.0f);
        gemm(stream, OUT_F32, false, h16, FF, 0, 0, eW2T + wEF, FF, 0, 0,
             tmp, E, 0, 0, nullptr, 0, 0, 0, ec2 + wE, M, E, FF, 1, 1, 1.0f);
        add_ln_kernel<<<dim3(M), dim3(256), 0, stream>>>(tmp, v1, eg2 + wE, ed2 + wE, x, x16);
    }
    hipMemcpyAsync(enc16, x16, BSE * sizeof(_Float16), hipMemcpyDeviceToDevice, stream);

    // ---------------- decoder ----------------
    cvt16_kernel<<<dim3((unsigned)(BSE / 1024)), dim3(256), 0, stream>>>(trg, x16);
    for (int l = 0; l < Lnum; ++l) {
        const size_t wEE = (size_t)l * EE, wE = (size_t)l * E;
        const size_t wEF = (size_t)l * EF, wF = (size_t)l * FF;
        const float* xres = (l == 0) ? trg : x;

        // self attention; v = LN(t + mha(t,t,t), g1, d1)
        attention_block(stream, x16, x16,
                        dWqT + wEE, dbq + wE, dWkT + wEE, dbk + wE,
                        dWvT + wEE, dbv + wE, dWoT + wEE, dbo + wE,
                        q16, k16, vT16, sc, p16, attn16, tmp);
        add_ln_kernel<<<dim3(M), dim3(256), 0, stream>>>(xres, tmp, dg1 + wE, dd1 + wE, v1, v116);

        // cross attention: queries from layer INPUT t; K/V from enc_out
        attention_block(stream, x16, enc16,
                        dWq2T + wEE, dbq2 + wE, dWk2T + wEE, dbk2 + wE,
                        dWv2T + wEE, dbv2 + wE, dWo2T + wEE, dbo2 + wE,
                        q16, k16, vT16, sc, p16, attn16, tmp);
        add_ln_kernel<<<dim3(M), dim3(256), 0, stream>>>(v1, tmp, dg2 + wE, dd2 + wE, b2, b216);

        gemm(stream, OUT_F16, true, b216, E, 0, 0, dW1T + wEF, E, 0, 0,
             nullptr, 0, 0, 0, h16, FF, 0, 0, dc1 + wF, M, FF, E, 1, 1, 1.0f);
        gemm(stream, OUT_F32, false, h16, FF, 0, 0, dW2T + wEF, FF, 0, 0,
             tmp, E, 0, 0, nullptr, 0, 0, 0, dc2 + wE, M, E, FF, 1, 1, 1.0f);
        add_ln_kernel<<<dim3(M), dim3(256), 0, stream>>>(b2, tmp, dg3 + wE, dd3 + wE, x, x16);
    }

    // ---------------- output projection: x16 @ WoutT^T + bout -> f32 --------
    gemm(stream, OUT_F32, false, x16, E, 0, 0, WoutT, E, 0, 0,
         (float*)d_out, Vv, 0, 0, nullptr, 0, 0, 0, bout, M, Vv, E, 1, 1, 1.0f);
}
